// MogrifierLSTMModel_33097017983725
// MI455X (gfx1250) — matrix operand order repaired and checked by reading, not compiled, not run on hardware
//
#include <hip/hip_runtime.h>

// ---------------------------------------------------------------------------
// Mogrifier-LSTM for MI455X (gfx1250, wave32, WMMA).
// All matmuls are Y = X @ W.T with W(out,in) row-major -> pre-transpose W to
// K-major bf16 once; every GEMM tile is v_wmma_f32_16x16x32_bf16.
// Recurrence: ONE persistent kernel (32 blocks) with software grid barriers
// between the 5 dependent phases per step (replaces 1280 tiny launches).
// Decoder: one deferred 16384x10000x1024 bf16 WMMA GEMM (h history = 32 MB,
// lives in the 192 MB L2).
// ---------------------------------------------------------------------------

#define SEQ    256
#define BATCH  64
#define VOC    10000
#define DIM    1024
#define HID    1024
#define G4     4096   // 4*HID
#define NBLK   32     // persistent-kernel blocks (8 waves each = 256 wave-tiles)

typedef __attribute__((ext_vector_type(16))) __bf16 v16bf;
typedef __attribute__((ext_vector_type(8)))  float  v8f;

union Frag { v16bf v; unsigned int u[8]; };
union Acc  { v8f   v; float        f[8]; };

__device__ __forceinline__ __bf16 f2bf(float f) {
  unsigned u = __float_as_uint(f);
  u += 0x7FFFu + ((u >> 16) & 1u);           // round-to-nearest-even
  unsigned short s = (unsigned short)(u >> 16);
  __bf16 h; __builtin_memcpy(&h, &s, 2); return h;
}
__device__ __forceinline__ float bf2f(__bf16 h) {
  unsigned short s; __builtin_memcpy(&s, &h, 2);
  return __uint_as_float((unsigned)s << 16);
}
__device__ __forceinline__ float sigm(float x) { return 1.0f / (1.0f + __expf(-x)); }

// A fragment: 16x32 bf16 tile of row-major A[lda], origin (row0,k0).
// ISA 7.12.2 16-bit A layout: lanes 0-15 -> M, half selects K-octet,
// VGPR j in 0..3 -> K = 8*half + 2j, j in 4..7 -> K = 16 + 8*half + 2(j-4).
__device__ __forceinline__ v16bf load_a(const __bf16* A, int lda,
                                        int row0, int k0) {
  unsigned lane = threadIdx.x & 31u;
  unsigned half = lane >> 4, m = lane & 15u;
  const __bf16* p = A + (size_t)(row0 + m) * lda + k0 + 8 * half;
  Frag f;
  f.u[0] = *(const unsigned int*)(p + 0);
  f.u[1] = *(const unsigned int*)(p + 2);
  f.u[2] = *(const unsigned int*)(p + 4);
  f.u[3] = *(const unsigned int*)(p + 6);
  f.u[4] = *(const unsigned int*)(p + 16);
  f.u[5] = *(const unsigned int*)(p + 18);
  f.u[6] = *(const unsigned int*)(p + 20);
  f.u[7] = *(const unsigned int*)(p + 22);
  return f.v;
}

// B fragment: 32x16 tile of K-major Bt[K rows, ldb cols], origin (k0,n0).
// Lane l -> column n0+(l&15); element s<8 -> K = 8*half+s, s>=8 -> K = 16+8*half+(s-8).
__device__ __forceinline__ v16bf load_b(const __bf16* Bt, int ldb,
                                        int k0, int n0) {
  unsigned lane = threadIdx.x & 31u;
  unsigned half = lane >> 4, q = lane & 15u;
  const __bf16* p = Bt + (size_t)(k0 + 8 * half) * ldb + n0 + q;
  Frag f;
#pragma unroll
  for (int j = 0; j < 8; ++j) {
    f.v[j] = p[(size_t)j * ldb];
    f.v[8 + j] = p[(size_t)(16 + j) * ldb];
  }
  return f.v;
}

__device__ __forceinline__ v8f wmma_bf16(v16bf a, v16bf b, v8f c) {
  return __builtin_amdgcn_wmma_f32_16x16x32_bf16(false, a, false, b,
                                                 (short)0, c, false, false);
}

// Software grid barrier for the persistent recurrence kernel.
// Monotonic counter (no reset races); release fence flushes this wave's
// stores, acquire fence after release invalidates per-CU L0 so every wave
// sees other WGPs' writes (agent scope -> global_inv on gfx1250).
__device__ __forceinline__ void gbar(unsigned* cnt, unsigned idx) {
  __builtin_amdgcn_fence(__ATOMIC_RELEASE, "agent");
  __syncthreads();
  if (threadIdx.x == 0) {
    __hip_atomic_fetch_add(cnt, 1u, __ATOMIC_RELAXED, __HIP_MEMORY_SCOPE_AGENT);
    while (__hip_atomic_load(cnt, __ATOMIC_RELAXED, __HIP_MEMORY_SCOPE_AGENT)
           < idx * (unsigned)NBLK)
      __builtin_amdgcn_s_sleep(2);
  }
  __syncthreads();
  __builtin_amdgcn_fence(__ATOMIC_ACQUIRE, "agent");
}

// ------------------------- one-time prep kernels ---------------------------

// W(out,in) f32 row-major -> Wt(in,out) bf16 K-major. Coalesced writes.
__global__ void k_transpose_bf16(const float* __restrict__ W,
                                 __bf16* __restrict__ Wt,
                                 int out, int in) {
  size_t total = (size_t)out * in;
  for (size_t idx = (size_t)blockIdx.x * blockDim.x + threadIdx.x; idx < total;
       idx += (size_t)gridDim.x * blockDim.x) {
    size_t n = idx % (size_t)out, k = idx / (size_t)out;
    Wt[idx] = f2bf(W[n * (size_t)in + k]);
  }
}

// Gather all embeddings up front: Xall[t,b,:] = bf16(emb[tok[t,b],:])
__global__ void k_embed(const int* __restrict__ tokens,
                        const float* __restrict__ emb,
                        __bf16* __restrict__ Xall) {
  size_t total = (size_t)SEQ * BATCH * DIM;
  for (size_t idx = (size_t)blockIdx.x * blockDim.x + threadIdx.x; idx < total;
       idx += (size_t)gridDim.x * blockDim.x) {
    size_t d = idx & (DIM - 1);
    size_t tb = idx >> 10;                    // t*BATCH + b
    int tok = tokens[tb];
    Xall[idx] = f2bf(emb[(size_t)tok * DIM + d]);
  }
}

__global__ void k_init(const float* __restrict__ h0, const float* __restrict__ c0,
                       __bf16* __restrict__ hcur, float* __restrict__ hF,
                       float* __restrict__ cF, unsigned* __restrict__ cnt) {
  int i = blockIdx.x * blockDim.x + threadIdx.x;
  if (i == 0) *cnt = 0u;                     // reset barrier counter every call
  if (i < BATCH * HID) { hcur[i] = f2bf(h0[i]); hF[i] = h0[i]; cF[i] = c0[i]; }
}

// ------------------------- recurrence phase bodies -------------------------

// P = A(64x1024) @ Wt(1024x1024); tgt[m,n] *= 2*sigmoid(P[m,n]+bias[n]) (bf16)
// 256 waves: 4 M-tiles x 64 N-tiles, K=1024 -> 32 WMMAs per wave.
__device__ __forceinline__ void mog_phase(const __bf16* A, const __bf16* Wt,
                                          const float* bias, __bf16* tgt, int w) {
  int mt = w & 3, nt = w >> 2;
  int row0 = mt * 16, n0 = nt * 16;
  Acc acc;
#pragma unroll
  for (int i = 0; i < 8; ++i) acc.f[i] = 0.0f;
  for (int k0 = 0; k0 < 1024; k0 += 32) {
    v16bf a = load_a(A, 1024, row0, k0);
    v16bf b = load_b(Wt, 1024, k0, n0);
    acc.v = wmma_bf16(a, b, acc.v);
  }
  unsigned lane = threadIdx.x & 31u;
  unsigned halfw = lane >> 4, q = lane & 15u;
  int ncol = n0 + (int)q;
  float bi = bias[ncol];
#pragma unroll
  for (int r = 0; r < 8; ++r) {
    int mrow = row0 + r + 8 * (int)halfw;
    size_t off = (size_t)mrow * 1024 + ncol;
    float g = 2.0f * sigm(acc.f[r] + bi);
    tgt[off] = f2bf(g * bf2f(tgt[off]));
  }
}

// Fused gates + LSTM cell: each wave owns one (m,n) tile of H, computes its
// i/f/g/o tiles: 4 accs x (x@Wih + h@Whh) = 256 WMMAs/wave.
__device__ __forceinline__ void cell_phase(const __bf16* x, const __bf16* h,
                                           const __bf16* wih_t, const __bf16* whh_t,
                                           const float* b_ih, const float* b_hh,
                                           float* cF, float* hF,
                                           __bf16* h_next, __bf16* Hrow, int w) {
  int mt = w & 3, nt = w >> 2;
  int row0 = mt * 16, n0 = nt * 16;
  Acc acc[4];
#pragma unroll
  for (int g = 0; g < 4; ++g)
#pragma unroll
    for (int i = 0; i < 8; ++i) acc[g].f[i] = 0.0f;

  for (int k0 = 0; k0 < 1024; k0 += 32) {
    v16bf ax = load_a(x, DIM, row0, k0);
    v16bf ah = load_a(h, HID, row0, k0);
#pragma unroll
    for (int g = 0; g < 4; ++g) {
      v16bf bi = load_b(wih_t, G4, k0, g * HID + n0);
      acc[g].v = wmma_bf16(ax, bi, acc[g].v);
      v16bf bh = load_b(whh_t, G4, k0, g * HID + n0);
      acc[g].v = wmma_bf16(ah, bh, acc[g].v);
    }
  }
  unsigned lane = threadIdx.x & 31u;
  unsigned halfw = lane >> 4, q = lane & 15u;
  int ncol = n0 + (int)q;
  float bi_ = b_ih[0 * HID + ncol] + b_hh[0 * HID + ncol];
  float bf_ = b_ih[1 * HID + ncol] + b_hh[1 * HID + ncol];
  float bg_ = b_ih[2 * HID + ncol] + b_hh[2 * HID + ncol];
  float bo_ = b_ih[3 * HID + ncol] + b_hh[3 * HID + ncol];
#pragma unroll
  for (int r = 0; r < 8; ++r) {
    int mrow = row0 + r + 8 * (int)halfw;
    size_t off = (size_t)mrow * HID + ncol;
    float gi = sigm(acc[0].f[r] + bi_);
    float gf = sigm(acc[1].f[r] + bf_);
    float gg = tanhf(acc[2].f[r] + bg_);
    float go = sigm(acc[3].f[r] + bo_);
    float cn = gf * cF[off] + gi * gg;
    float hn = go * tanhf(cn);
    cF[off] = cn;
    hF[off] = hn;
    __bf16 hb = f2bf(hn);
    h_next[off] = hb;
    Hrow[off] = hb;
  }
}

// Persistent recurrence kernel: all 256 steps, grid-synced phases.
// 32 blocks x 256 threads (trivially co-resident on MI455X).
__global__ void __launch_bounds__(256) k_recurrence(
    __bf16* Xall, const __bf16* qwt, const float* qb,
    const __bf16* rwt, const float* rb,
    const __bf16* wiht, const __bf16* whht,
    const float* b_ih, const float* b_hh,
    __bf16* hb0, __bf16* hb1, float* cF, float* hF,
    __bf16* Hbuf, unsigned* cnt) {
  int w = blockIdx.x * 8 + (threadIdx.x >> 5);
  unsigned bar = 0;
  for (int t = 0; t < SEQ; ++t) {
    __bf16* xt   = Xall + (size_t)t * BATCH * DIM;
    __bf16* hin  = (t & 1) ? hb1 : hb0;
    __bf16* hout = (t & 1) ? hb0 : hb1;
    for (int r = 0; r < 2; ++r) {
      mog_phase(hin, qwt + (size_t)r * HID * DIM, qb + (size_t)r * DIM, xt, w);
      gbar(cnt, ++bar);
      mog_phase(xt, rwt + (size_t)r * DIM * HID, rb + (size_t)r * HID, hin, w);
      gbar(cnt, ++bar);
    }
    cell_phase(xt, hin, wiht, whht, b_ih, b_hh, cF, hF, hout,
               Hbuf + (size_t)t * BATCH * HID, w);
    gbar(cnt, ++bar);
  }
}

// ------------------------------ decoder GEMM -------------------------------
// (SEQ*B=16384, K=1024) x (1024, V=10000). 1024 M-tiles x 625 N-tiles,
// 4-wide N register blocking per wave (A-frag reuse). Branches are
// wave-uniform so EXEC stays all-ones around the WMMAs.
__global__ void __launch_bounds__(256) k_decode(const __bf16* __restrict__ Hbuf,
                                                const __bf16* __restrict__ dwt,
                                                const float* __restrict__ db,
                                                float* __restrict__ out) {
  int wid = blockIdx.x * 8 + (threadIdx.x >> 5);
  int mt = wid & 1023;
  int ng = wid >> 10;                        // 0..156, 4 N-tiles each
  int row0 = mt * 16;
  Acc acc[4];
#pragma unroll
  for (int s = 0; s < 4; ++s)
#pragma unroll
    for (int i = 0; i < 8; ++i) acc[s].f[i] = 0.0f;

  for (int k0 = 0; k0 < 1024; k0 += 32) {
    v16bf a = load_a(Hbuf, HID, row0, k0);
#pragma unroll
    for (int s = 0; s < 4; ++s) {
      int ntile = ng * 4 + s;
      if (ntile < 625) {                     // wave-uniform guard
        v16bf b = load_b(dwt, VOC, k0, ntile * 16);
        acc[s].v = wmma_bf16(a, b, acc[s].v);
      }
    }
  }
  unsigned lane = threadIdx.x & 31u;
  unsigned halfw = lane >> 4, q = lane & 15u;
#pragma unroll
  for (int s = 0; s < 4; ++s) {
    int ntile = ng * 4 + s;
    if (ntile >= 625) continue;
    int ncol = ntile * 16 + (int)q;
    float bias = db[ncol];
#pragma unroll
    for (int r = 0; r < 8; ++r) {
      int mrow = row0 + r + 8 * (int)halfw;
      out[(size_t)mrow * VOC + ncol] = acc[s].f[r] + bias;
    }
  }
}

__global__ void k_tail(const float* __restrict__ hF, const float* __restrict__ cF,
                       float* __restrict__ out) {
  int i = blockIdx.x * blockDim.x + threadIdx.x;
  if (i < BATCH * HID) { out[i] = hF[i]; out[BATCH * HID + i] = cF[i]; }
}

// ------------------------------- launcher ----------------------------------

extern "C" void kernel_launch(void* const* d_in, const int* in_sizes, int n_in,
                              void* d_out, int out_size, void* d_ws, size_t ws_size,
                              hipStream_t stream) {
  (void)in_sizes; (void)n_in; (void)out_size; (void)ws_size;
  const int*   tokens = (const int*)  d_in[0];
  const float* emb    = (const float*)d_in[1];
  const float* qw     = (const float*)d_in[2];
  const float* qb     = (const float*)d_in[3];
  const float* rw     = (const float*)d_in[4];
  const float* rb     = (const float*)d_in[5];
  const float* w_ih   = (const float*)d_in[6];
  const float* w_hh   = (const float*)d_in[7];
  const float* b_ih   = (const float*)d_in[8];
  const float* b_hh   = (const float*)d_in[9];
  const float* dec_w  = (const float*)d_in[10];
  const float* dec_b  = (const float*)d_in[11];
  const float* h0     = (const float*)d_in[12];
  const float* c0     = (const float*)d_in[13];
  float* out = (float*)d_out;

  // Bump-allocate workspace (~112 MB total).
  char* p = (char*)d_ws;
  auto alloc = [&](size_t bytes) -> char* {
    char* r = p; p += (bytes + 255) & ~(size_t)255; return r;
  };
  __bf16* qwt  = (__bf16*)alloc(2ull * HID * DIM * 2);       // per-r K-major
  __bf16* rwt  = (__bf16*)alloc(2ull * DIM * HID * 2);
  __bf16* wiht = (__bf16*)alloc((size_t)DIM * G4 * 2);
  __bf16* whht = (__bf16*)alloc((size_t)HID * G4 * 2);
  __bf16* dwt  = (__bf16*)alloc((size_t)HID * VOC * 2);
  __bf16* Xall = (__bf16*)alloc((size_t)SEQ * BATCH * DIM * 2);
  __bf16* Hbuf = (__bf16*)alloc((size_t)SEQ * BATCH * HID * 2);
  __bf16* hb0  = (__bf16*)alloc((size_t)BATCH * HID * 2);
  __bf16* hb1  = (__bf16*)alloc((size_t)BATCH * HID * 2);
  float*  hF   = (float*) alloc((size_t)BATCH * HID * 4);
  float*  cF   = (float*) alloc((size_t)BATCH * HID * 4);
  unsigned* cnt = (unsigned*)alloc(256);

  // One-time conversions (re-done every call: deterministic, no cached state).
  for (int r = 0; r < 2; ++r) {
    k_transpose_bf16<<<4096, 256, 0, stream>>>(qw + (size_t)r * DIM * HID,
                                               qwt + (size_t)r * HID * DIM, DIM, HID);
    k_transpose_bf16<<<4096, 256, 0, stream>>>(rw + (size_t)r * HID * DIM,
                                               rwt + (size_t)r * DIM * HID, HID, DIM);
  }
  k_transpose_bf16<<<16384, 256, 0, stream>>>(w_ih, wiht, G4, DIM);
  k_transpose_bf16<<<16384, 256, 0, stream>>>(w_hh, whht, G4, HID);
  k_transpose_bf16<<<40000, 256, 0, stream>>>(dec_w, dwt, VOC, HID);
  k_embed<<<32768, 256, 0, stream>>>(tokens, emb, Xall);
  k_init<<<256, 256, 0, stream>>>(h0, c0, hb0, hF, cF, cnt);

  // Whole 256-step recurrence in one persistent, grid-synced kernel.
  k_recurrence<<<NBLK, 256, 0, stream>>>(Xall, qwt, qb, rwt, rb, wiht, whht,
                                         b_ih, b_hh, hb0, hb1, cF, hF,
                                         Hbuf, cnt);

  // Deferred decoder GEMM over the whole h history + output tail.
  k_decode<<<20096, 256, 0, stream>>>(Hbuf, dwt, dec_b, out);
  k_tail<<<256, 256, 0, stream>>>(hF, cF, out + (size_t)SEQ * BATCH * VOC);
}
